// HalutMatmul_49684181680224
// MI455X (gfx1250) — matrix operand order repaired and checked by reading, not compiled, not run on hardware
//
#include <hip/hip_runtime.h>

typedef __attribute__((ext_vector_type(16))) _Float16 v16h;
typedef __attribute__((ext_vector_type(8)))  float    v8f;

#define CB      16      // codebooks
#define DEPTH   4
#define DSUB    9       // dims per codebook subspace (D = 144)
#define KCODES  16
#define MOUT    64
#define NNODES  15
#define LELEMS  (MOUT * CB * KCODES)   // 16384

// ---------------------------------------------------------------------------
// Kernel 1: split L (f32, layout [m][c][k]) into f16 hi + f16 lo residual in
// d_ws, relaid out as [c][m][k] so each WMMA B-fragment lane reads 32
// contiguous bytes.  hi at offset 0, lo at offset LELEMS halves.
// ---------------------------------------------------------------------------
__global__ void halut_convert_L(const float* __restrict__ L,
                                _Float16* __restrict__ dst) {
    int idx = blockIdx.x * blockDim.x + threadIdx.x;
    if (idx >= LELEMS) return;
    int k = idx & 15;
    int c = (idx >> 4) & 15;
    int m = idx >> 8;
    float v = L[idx];
    _Float16 hi = (_Float16)v;
    _Float16 lo = (_Float16)(v - (float)hi);
    int didx = (((c << 6) + m) << 4) + k;      // ((c*64 + m)*16 + k)
    dst[didx]          = hi;
    dst[LELEMS + didx] = lo;
}

// ---------------------------------------------------------------------------
// Kernel 2: fused encode + WMMA decode.
//   Phase 1 (per thread, f32): vectorized b128 row loads, hash 16 codebooks,
//            depth-4 tree traversal, pack 16 x 4-bit codes -> LDS.
//   Phase 2 (per wave): one-hot(f16) x L_{hi,lo}(f16) via
//            v_wmma_f32_16x16x32_f16, f32 accumulate.  Loop nest keeps each
//            A-fragment live only across the 4 M-tiles (8 persistent v8f
//            accumulators) so VGPR use stays well under 256.
// ---------------------------------------------------------------------------
__global__ __launch_bounds__(256) void halut_main(
    const float* __restrict__ I,
    const float* __restrict__ A,
    const float* __restrict__ T,
    const _Float16* __restrict__ Lhl,   // hi at 0, lo at LELEMS
    float* __restrict__ out)
{
    __shared__ float        Alds[CB * DSUB * DEPTH];   // 576 f32
    __shared__ float        Tlds[CB * NNODES];         // 240 f32
    __shared__ unsigned int codesLds[256 * 2];         // 16 x 4-bit codes / row

    const int t = threadIdx.x;

    for (int i = t; i < CB * DSUB * DEPTH; i += 256) Alds[i] = A[i];
    for (int i = t; i < CB * NNODES;       i += 256) Tlds[i] = T[i];
    __syncthreads();

    // ---------------- Phase 1: per-row encode (f32, exact signs) ----------
    const long long row  = (long long)blockIdx.x * 256 + t;
    const float* Irow = I + row * (CB * DSUB);   // 576 B row stride: 16B aligned

    unsigned int p0 = 0u, p1 = 0u;
    #pragma unroll
    for (int g = 0; g < 4; ++g) {                // 4 codebooks per group
        // 9 x float4 = 36 floats = codebooks 4g..4g+3 (16B-aligned b128 loads)
        float xs[36];
        const float4* Ip = (const float4*)(Irow + g * 36);
        #pragma unroll
        for (int q = 0; q < 9; ++q) {
            float4 v = Ip[q];
            xs[4 * q + 0] = v.x; xs[4 * q + 1] = v.y;
            xs[4 * q + 2] = v.z; xs[4 * q + 3] = v.w;
        }
        #pragma unroll
        for (int cc = 0; cc < 4; ++cc) {
            const int c = g * 4 + cc;
            float h0 = 0.f, h1 = 0.f, h2 = 0.f, h3 = 0.f;
            const float* Ac = Alds + c * (DSUB * DEPTH);
            #pragma unroll
            for (int j = 0; j < DSUB; ++j) {
                float x = xs[cc * DSUB + j];
                h0 = fmaf(x, Ac[j * 4 + 0], h0);
                h1 = fmaf(x, Ac[j * 4 + 1], h1);
                h2 = fmaf(x, Ac[j * 4 + 2], h2);
                h3 = fmaf(x, Ac[j * 4 + 3], h3);
            }
            const float* Tc = Tlds + c * NNODES;
            unsigned pre = (h0 > Tc[0]) ? 1u : 0u;             // lvl0: node 0
            pre = 2u * pre + ((h1 > Tc[1 + pre]) ? 1u : 0u);   // lvl1: 1+pre
            pre = 2u * pre + ((h2 > Tc[3 + pre]) ? 1u : 0u);   // lvl2: 3+pre
            pre = 2u * pre + ((h3 > Tc[7 + pre]) ? 1u : 0u);   // lvl3: 7+pre
            if (c < 8) p0 |= pre << (4 * c);
            else       p1 |= pre << (4 * (c - 8));
        }
    }
    codesLds[2 * t + 0] = p0;
    codesLds[2 * t + 1] = p1;
    __syncthreads();

    // ---------------- Phase 2: WMMA decode (one-hot @ L_hi|L_lo) ----------
    const int lane   = t & 31;
    const int wv     = t >> 5;
    const int lm     = lane & 15;            // row-in-tile (A) / N column (B,D)
    const int hiHalf = (lane >> 4) & 1;      // 0: K 0..15 (hi), 1: K 16..31 (lo)
    const int kbase  = hiHalf << 3;          // A one-hot K base (per layout)

    // packed codes for this wave's two 16-row tiles
    const unsigned q00 = codesLds[2 * (wv * 32 +      lm) + 0];
    const unsigned q01 = codesLds[2 * (wv * 32 +      lm) + 1];
    const unsigned q10 = codesLds[2 * (wv * 32 + 16 + lm) + 0];
    const unsigned q11 = codesLds[2 * (wv * 32 + 16 + lm) + 1];

    const long long waveRow = (long long)blockIdx.x * 256 + wv * 32;
    const _Float16* Lsel = Lhl + (hiHalf ? LELEMS : 0);   // lanes>=16 read lo
    const _Float16* Lcol = Lsel + (lm << 4);              // + m-within-tile

    // acc[2*mt + tile]: 8 x v8f = 64 VGPRs, persistent
    v8f acc[8];
    #pragma unroll
    for (int i = 0; i < 8; ++i) acc[i] = (v8f){};

    #pragma unroll
    for (int c = 0; c < CB; ++c) {
        const unsigned k0 = (c < 8) ? ((q00 >> (4 * c)) & 15u)
                                    : ((q01 >> (4 * (c - 8))) & 15u);
        const unsigned k1 = (c < 8) ? ((q10 >> (4 * c)) & 15u)
                                    : ((q11 >> (4 * (c - 8))) & 15u);
        // one-hot A fragments; duplicate into K+16 to also select L_lo
        v16h a0, a1;
        #pragma unroll
        for (int j = 0; j < 16; ++j) {
            _Float16 e0 = (k0 == (unsigned)j) ? (_Float16)1.0f
                                                        : (_Float16)0.0f;
            _Float16 e1 = (k1 == (unsigned)j) ? (_Float16)1.0f
                                                        : (_Float16)0.0f;
            a0[j] = e0;
            a1[j] = e1;
        }
        // consume this codebook's fragment across all 4 M-tiles immediately
        #pragma unroll
        for (int mt = 0; mt < 4; ++mt) {
            const v16h b = *(const v16h*)(Lcol + (((c << 6) + (mt << 4)) << 4));
            acc[2 * mt + 0] = __builtin_amdgcn_wmma_f32_16x16x32_f16(
                false, a0, false, b, (short)0, acc[2 * mt + 0], false, false);
            acc[2 * mt + 1] = __builtin_amdgcn_wmma_f32_16x16x32_f16(
                false, a1, false, b, (short)0, acc[2 * mt + 1], false, false);
        }
    }

    // D layout: elem i -> M = i + 8*hiHalf, N = lm
    #pragma unroll
    for (int mt = 0; mt < 4; ++mt) {
        #pragma unroll
        for (int i = 0; i < 8; ++i) {
            const long long r0 = waveRow +      (hiHalf ? 8 : 0) + i;
            const long long r1 = waveRow + 16 + (hiHalf ? 8 : 0) + i;
            out[r0 * MOUT + mt * 16 + lm] = acc[2 * mt + 0][i];
            out[r1 * MOUT + mt * 16 + lm] = acc[2 * mt + 1][i];
        }
    }
}

// ---------------------------------------------------------------------------
extern "C" void kernel_launch(void* const* d_in, const int* in_sizes, int n_in,
                              void* d_out, int out_size, void* d_ws,
                              size_t ws_size, hipStream_t stream) {
    (void)n_in; (void)out_size; (void)ws_size;
    const float* I = (const float*)d_in[0];
    const float* A = (const float*)d_in[1];
    const float* T = (const float*)d_in[2];
    const float* L = (const float*)d_in[3];
    // d_in[4] (S) and d_in[5] (B) are encoded implicitly by the tree traversal.
    float*     out = (float*)d_out;
    _Float16*  Lhl = (_Float16*)d_ws;      // needs 64 KB scratch

    const long long N = (long long)in_sizes[0] / (CB * DSUB);  // rows

    halut_convert_L<<<(LELEMS + 255) / 256, 256, 0, stream>>>(L, Lhl);

    const int blocks = (int)(N / 256);     // N = 131072 -> 512 blocks
    halut_main<<<blocks, 256, 0, stream>>>(I, A, T, Lhl, out);
}
